// rGIN_78683800863473
// MI455X (gfx1250) — matrix operand order repaired and checked by reading, not compiled, not run on hardware
//
#include <hip/hip_runtime.h>
#include <hip/hip_bf16.h>
#include <stdint.h>

// ---------------- problem constants (match reference) ----------------
#define GN    50000      // nodes
#define GE    800000     // edges
#define GD    128        // in channels
#define GH    128        // hidden
#define GL    4          // layers
#define HS    160        // padded feature stride (multiple of 32 for WMMA K tiles)
#define BN_EPS 1e-5f

// ---------------- WMMA vector types ----------------
typedef __attribute__((ext_vector_type(16))) _Float16 v16h;
typedef __attribute__((ext_vector_type(8)))  _Float16 v8h;
typedef __attribute__((ext_vector_type(8)))  float    v8f;

union V16u { v16h v; v8h h[2]; };

// ---------------- atomic helpers ----------------
__device__ __forceinline__ void atomicAddGF(float* p, float v) {
    __hip_atomic_fetch_add(p, v, __ATOMIC_RELAXED, __HIP_MEMORY_SCOPE_AGENT);
}
__device__ __forceinline__ void atomicAddLF(float* p, float v) {
    __hip_atomic_fetch_add(p, v, __ATOMIC_RELAXED, __HIP_MEMORY_SCOPE_WORKGROUP);
}

// ---------------- CDNA5 async global->LDS (ASYNCcnt) with safe fallback ----------
#if __has_builtin(__builtin_amdgcn_global_load_async_to_lds_b128)
#define HAVE_ASYNC_LDS 1
// builtin prototype (from hipcc diagnostic): (int4 AS1* src, int4 AS3* dst, Ii off, Ii cpol)
typedef int v4i_vs __attribute__((vector_size(16)));
typedef __attribute__((address_space(1))) v4i_vs* gptr_b128;
typedef __attribute__((address_space(3))) v4i_vs* lptr_b128;
__device__ __forceinline__ void async_copy16(const void* g, void* l) {
    // generic->AS1 keeps the 64-bit value; generic LDS addr low 32 bits = LDS offset
    __builtin_amdgcn_global_load_async_to_lds_b128(
        (gptr_b128)(uintptr_t)g,
        (lptr_b128)(uint32_t)(uintptr_t)l,
        0, 0);
}
__device__ __forceinline__ void wait_async0() {
#if __has_builtin(__builtin_amdgcn_s_wait_asynccnt)
    __builtin_amdgcn_s_wait_asynccnt(0);
#else
    asm volatile("s_wait_asynccnt 0x0" ::: "memory");
#endif
}
#else
#define HAVE_ASYNC_LDS 0
#endif

// ---------------- kernels ----------------

// h0[row, 0:128] = x[row], h0[row,128] = r[row], h0[row,129:160] = 0
__global__ void build_h0_kernel(const float* __restrict__ x,
                                const float* __restrict__ r,
                                float* __restrict__ h) {
    int idx = blockIdx.x * 256 + threadIdx.x;
    if (idx >= GN * HS) return;
    int row = idx / HS, c = idx - row * HS;
    float v = 0.0f;
    if (c < GD)       v = x[(size_t)row * GD + c];
    else if (c == GD) v = r[row];
    h[idx] = v;
}

// full-stride float4 copy: z = h (used once before layer 0)
__global__ void copy_kernel(const float* __restrict__ src, float* __restrict__ dst, int count4) {
    int idx = blockIdx.x * 256 + threadIdx.x;
    if (idx >= count4) return;
    reinterpret_cast<float4*>(dst)[idx] = reinterpret_cast<const float4*>(src)[idx];
}

// scatter-add: z[dst] += h[src] over edges; one wave per edge, coalesced lanes.
// ~102M L2-resident f32 atomics per layer -> the roofline-dominant term.
__global__ void agg_kernel(const int* __restrict__ ei,
                           const float* __restrict__ h,
                           float* __restrict__ z, int ncols) {
    int tid = blockIdx.x * 256 + threadIdx.x;
    int e = tid >> 5;
    int lane = tid & 31;
    if (e >= GE) return;
    int src = ei[e];
    int dst = ei[GE + e];
    const float* hp = h + (size_t)src * HS;
    float*       zp = z + (size_t)dst * HS;
#pragma unroll
    for (int i = 0; i < 5; ++i) {
        int c = lane + i * 32;
        if (c < ncols) atomicAddGF(zp + c, hp[c]);
    }
}

// W (din x 128, f32, row-major) -> Wt (128 x Kpad, f16, transposed, zero-padded K)
__global__ void convw_kernel(const float* __restrict__ W, _Float16* __restrict__ Wt,
                             int din, int Kpad) {
    int idx = blockIdx.x * 256 + threadIdx.x;
    if (idx >= GH * Kpad) return;
    int n = idx / Kpad, k = idx - n * Kpad;
    float v = (k < din) ? W[(size_t)k * GH + n] : 0.0f;
    Wt[idx] = (_Float16)v;
}

// C = A(f32, lda) * Wt^T(f16) + bias ; per-column sum/sumsq accumulated for BN.
// One 128x128 C tile per 256-thread (8-wave) block; each wave owns 32x64 = 8
// WMMA 16x16x32 tiles. Full weight matrix LDS-resident (<=40KB, async-loaded);
// A tile double-buffered -> single barrier per K step.
#define LDSW 40    // A-tile LDS row stride (halfs): 80B, 16B-aligned, bank-friendly
#define BSW  168   // resident-B LDS row stride (halfs): 336B -> bank step 20 mod 64
__global__ __launch_bounds__(256) void gemm_bn_kernel(
    const float* __restrict__ A, int lda,
    const _Float16* __restrict__ Wt, int Kpad,
    const float* __restrict__ bias,
    float* __restrict__ T,
    float* __restrict__ gsum, float* __restrict__ gsumsq,
    int Ksteps, int Nrows) {

    __shared__ _Float16 Bs[128 * BSW];          // 43008 B
    __shared__ _Float16 As[2][128 * LDSW];      // 20480 B
    __shared__ float csum[GH], csumsq[GH], biasS[GH];

    const int t = threadIdx.x;
    if (t < GH) { csum[t] = 0.0f; csumsq[t] = 0.0f; biasS[t] = bias[t]; }

    const int rowBase = blockIdx.x * 128;
    const int w    = t >> 5;
    const int lane = t & 31;
    const int wm = w & 3;        // 0..3 : 32-row group
    const int wn = w >> 2;       // 0..1 : 64-col group
    const int lhalf = lane >> 4; // 0|1
    const int lmod  = lane & 15;

    // ---- preload entire transposed weight matrix into LDS ----
    const int rowChunks = Kpad >> 3;            // 16B chunks per weight row
    for (int idx = t; idx < 128 * rowChunks; idx += 256) {
        int nn = idx / rowChunks;
        int c8 = (idx - nn * rowChunks) * 8;
        const _Float16* gp = Wt + (size_t)nn * Kpad + c8;
        _Float16*       lp = &Bs[nn * BSW + c8];
#if HAVE_ASYNC_LDS
        async_copy16(gp, lp);
#else
        *reinterpret_cast<uint4*>(lp) = *reinterpret_cast<const uint4*>(gp);
#endif
    }

    v8f acc[2][4];
#pragma unroll
    for (int mi = 0; mi < 2; ++mi)
#pragma unroll
        for (int ni = 0; ni < 4; ++ni)
#pragma unroll
            for (int e = 0; e < 8; ++e) acc[mi][ni][e] = 0.0f;

    // ---- A tile stage: 128 rows x 32 k, f32 -> f16, b128 LDS stores ----
    auto stageA = [&](int ks, int buf) {
        const int k0 = ks * 32;
#pragma unroll
        for (int i = 0; i < 2; ++i) {
            int idx = t + i * 256;              // 0..511 slots of 8 halfs
            int rr = idx >> 2;
            int c8 = (idx & 3) * 8;
            int gr = rowBase + rr;
            float4 v0 = make_float4(0.f, 0.f, 0.f, 0.f);
            float4 v1 = v0;
            if (gr < Nrows) {
                const float* gp = A + (size_t)gr * lda + k0 + c8;
                v0 = *reinterpret_cast<const float4*>(gp);
                v1 = *reinterpret_cast<const float4*>(gp + 4);
            }
            v8h hv;
            hv[0] = (_Float16)v0.x; hv[1] = (_Float16)v0.y;
            hv[2] = (_Float16)v0.z; hv[3] = (_Float16)v0.w;
            hv[4] = (_Float16)v1.x; hv[5] = (_Float16)v1.y;
            hv[6] = (_Float16)v1.z; hv[7] = (_Float16)v1.w;
            *reinterpret_cast<v8h*>(&As[buf][rr * LDSW + c8]) = hv;
        }
    };

    stageA(0, 0);
#if HAVE_ASYNC_LDS
    wait_async0();
#endif
    __syncthreads();

    for (int ks = 0; ks < Ksteps; ++ks) {
        const int buf = ks & 1;
        if (ks + 1 < Ksteps) stageA(ks + 1, buf ^ 1);
        const int k0 = ks * 32;
#pragma unroll
        for (int mi = 0; mi < 2; ++mi) {
            const int arow = wm * 32 + mi * 16 + lmod;
            // A lane layout (ISA 7.12.2): two 8-half runs at k = base, base+16
            const _Float16* ap = &As[buf][arow * LDSW + lhalf * 8];
            V16u a;
            a.h[0] = *reinterpret_cast<const v8h*>(ap);
            a.h[1] = *reinterpret_cast<const v8h*>(ap + 16);
#pragma unroll
            for (int ni = 0; ni < 4; ++ni) {
                const int bcol = wn * 64 + ni * 16 + lmod;
                // B lane layout: two 8-half runs at k = k0 + lhalf*8, +16 (matches A)
                const _Float16* bp = &Bs[bcol * BSW + k0 + lhalf * 8];
                V16u b;
                b.h[0] = *reinterpret_cast<const v8h*>(bp);
                b.h[1] = *reinterpret_cast<const v8h*>(bp + 16);
                acc[mi][ni] = __builtin_amdgcn_wmma_f32_16x16x32_f16(
                    false, a.v, false, b.v, (short)0, acc[mi][ni], false, false);
            }
        }
        __syncthreads();
    }

    // ---- epilogue: bias add, store, per-column stats via LDS then global atomics ----
#pragma unroll
    for (int mi = 0; mi < 2; ++mi) {
#pragma unroll
        for (int ni = 0; ni < 4; ++ni) {
            const int col = wn * 64 + ni * 16 + lmod;
            const float bv = biasS[col];
            float s = 0.0f, s2 = 0.0f;
            const int rtile = rowBase + wm * 32 + mi * 16 + lhalf * 8;
#pragma unroll
            for (int r = 0; r < 8; ++r) {
                int gr = rtile + r;
                if (gr < Nrows) {
                    float v = acc[mi][ni][r] + bv;
                    T[(size_t)gr * GH + col] = v;
                    s += v; s2 += v * v;
                }
            }
            atomicAddLF(&csum[col], s);
            atomicAddLF(&csumsq[col], s2);
        }
    }
    __syncthreads();
    if (t < GH) {
        atomicAddGF(&gsum[t],   csum[t]);
        atomicAddGF(&gsumsq[t], csumsq[t]);
    }
}

// BN after linear1: scale/shift from batch stats + (g, be)
__global__ void bn1_prep_kernel(const float* __restrict__ sum, const float* __restrict__ sq,
                                const float* __restrict__ g, const float* __restrict__ be,
                                float* __restrict__ scale, float* __restrict__ shift) {
    int t = threadIdx.x;
    const float inv_n = 1.0f / (float)GN;
    float m = sum[t] * inv_n;
    float v = sq[t] * inv_n - m * m;
    float a = g[t] * rsqrtf(v + BN_EPS);
    scale[t] = a;
    shift[t] = be[t] - m * a;
}

// Composition of bn2 followed by bn_out (both on batch stats of t2):
//   y = a*t + (be2 - a*m),  mean_y = be2,  var_y = a^2 * v
//   out = S*t + (be_out - S*m)  with  S = g_out * rsqrt(a^2 v + eps) * a
__global__ void bn2_prep_kernel(const float* __restrict__ sum, const float* __restrict__ sq,
                                const float* __restrict__ g,  const float* __restrict__ be,
                                const float* __restrict__ gout, const float* __restrict__ beout,
                                float* __restrict__ scale, float* __restrict__ shift) {
    int t = threadIdx.x;
    (void)be; // inner shift cancels exactly under batch-stat renormalization
    const float inv_n = 1.0f / (float)GN;
    float m = sum[t] * inv_n;
    float v = sq[t] * inv_n - m * m;
    float a = g[t] * rsqrtf(v + BN_EPS);
    float c = rsqrtf(a * a * v + BN_EPS);
    float S = gout[t] * c * a;
    scale[t] = S;
    shift[t] = beout[t] - S * m;
}

// out[row,c] = (relu?)(T[row,c]*scale[c]+shift[c]); optional dual write (h and z
// accumulator for the next layer) to kill the standalone copy pass.
__global__ void apply_kernel(const float* __restrict__ T,
                             const float* __restrict__ scale, const float* __restrict__ shift,
                             float* __restrict__ out, int ostride,
                             float* __restrict__ out2, int relu) {
    int idx = blockIdx.x * 256 + threadIdx.x;           // over N*32
    if (idx >= GN * 32) return;
    int row = idx >> 5;
    int c4  = (idx & 31) << 2;
    float4 tv = *reinterpret_cast<const float4*>(T + (size_t)row * GH + c4);
    float4 sc = *reinterpret_cast<const float4*>(scale + c4);
    float4 sh = *reinterpret_cast<const float4*>(shift + c4);
    float4 o;
    o.x = fmaf(tv.x, sc.x, sh.x);
    o.y = fmaf(tv.y, sc.y, sh.y);
    o.z = fmaf(tv.z, sc.z, sh.z);
    o.w = fmaf(tv.w, sc.w, sh.w);
    if (relu) {
        o.x = fmaxf(o.x, 0.f); o.y = fmaxf(o.y, 0.f);
        o.z = fmaxf(o.z, 0.f); o.w = fmaxf(o.w, 0.f);
    }
    *reinterpret_cast<float4*>(out + (size_t)row * ostride + c4) = o;
    if (out2) *reinterpret_cast<float4*>(out2 + (size_t)row * HS + c4) = o;
}

// ---------------- host orchestration ----------------
extern "C" void kernel_launch(void* const* d_in, const int* in_sizes, int n_in,
                              void* d_out, int out_size, void* d_ws, size_t ws_size,
                              hipStream_t stream) {
    (void)in_sizes; (void)n_in; (void)out_size; (void)ws_size;

    const float* x  = (const float*)d_in[0];
    const int*   ei = (const int*)d_in[1];
    const float* r  = (const float*)d_in[2];
    // params: 10 tensors per layer starting at index 3:
    // W1,b1,g1,be1,W2,b2,g2,be2,g_out,be_out

    // workspace layout (floats)
    float* ws   = (float*)d_ws;
    float* Hbuf = ws;                                   // [GN x HS]
    float* Zbuf = Hbuf + (size_t)GN * HS;               // [GN x HS]
    float* Tbuf = Zbuf + (size_t)GN * HS;               // [GN x GH]
    _Float16* Wt = (_Float16*)(Tbuf + (size_t)GN * GH); // [128 x 160] max
    float* stats = (float*)(Wt + (size_t)GH * HS);      // sum[128], sq[128]
    float* scale = stats + 256;                         // [128]
    float* shift = scale + 128;                         // [128]

    const int copy4 = (GN * HS) / 4;
    dim3 blk(256);

    build_h0_kernel<<<(GN * HS + 255) / 256, blk, 0, stream>>>(x, r, Hbuf);
    copy_kernel<<<(copy4 + 255) / 256, blk, 0, stream>>>(Hbuf, Zbuf, copy4);

    for (int layer = 0; layer < GL; ++layer) {
        const int din  = (layer == 0) ? (GD + 1) : GH;
        const int Kpad = (layer == 0) ? HS : GH;      // 160 or 128
        const float* W1  = (const float*)d_in[3 + layer * 10 + 0];
        const float* b1  = (const float*)d_in[3 + layer * 10 + 1];
        const float* g1  = (const float*)d_in[3 + layer * 10 + 2];
        const float* be1 = (const float*)d_in[3 + layer * 10 + 3];
        const float* W2  = (const float*)d_in[3 + layer * 10 + 4];
        const float* b2  = (const float*)d_in[3 + layer * 10 + 5];
        const float* g2  = (const float*)d_in[3 + layer * 10 + 6];
        const float* be2 = (const float*)d_in[3 + layer * 10 + 7];
        const float* go  = (const float*)d_in[3 + layer * 10 + 8];
        const float* beo = (const float*)d_in[3 + layer * 10 + 9];

        // z (pre-initialized to h) += segment_sum(h[src] -> dst)
        agg_kernel<<<(GE * 32) / 256, blk, 0, stream>>>(ei, Hbuf, Zbuf, din);

        // linear1 + BN stats
        convw_kernel<<<(GH * Kpad + 255) / 256, blk, 0, stream>>>(W1, Wt, din, Kpad);
        (void)hipMemsetAsync(stats, 0, 256 * sizeof(float), stream);
        gemm_bn_kernel<<<(GN + 127) / 128, blk, 0, stream>>>(
            Zbuf, HS, Wt, Kpad, b1, Tbuf, stats, stats + 128, Kpad / 32, GN);
        bn1_prep_kernel<<<1, 128, 0, stream>>>(stats, stats + 128, g1, be1, scale, shift);
        apply_kernel<<<(GN * 32 + 255) / 256, blk, 0, stream>>>(
            Tbuf, scale, shift, Hbuf, HS, nullptr, /*relu=*/1);

        // linear2 + BN stats ; bn2 and bn_out compose into one affine
        convw_kernel<<<(GH * GH + 255) / 256, blk, 0, stream>>>(W2, Wt, GH, GH);
        (void)hipMemsetAsync(stats, 0, 256 * sizeof(float), stream);
        gemm_bn_kernel<<<(GN + 127) / 128, blk, 0, stream>>>(
            Hbuf, HS, Wt, GH, b2, Tbuf, stats, stats + 128, GH / 32, GN);
        bn2_prep_kernel<<<1, 128, 0, stream>>>(stats, stats + 128, g2, be2, go, beo, scale, shift);

        const int last = (layer == GL - 1);
        apply_kernel<<<(GN * 32 + 255) / 256, blk, 0, stream>>>(
            Tbuf, scale, shift,
            last ? (float*)d_out : Hbuf,
            last ? GH : HS,
            last ? nullptr : Zbuf,     // seed next layer's scatter accumulator
            /*relu=*/last ? 0 : 1);
    }
}
